// SelfJoinLayer_9320079032794
// MI455X (gfx1250) — compile-verified, matrix operand order checked by reading, not run on hardware
//
#include <hip/hip_runtime.h>
#include <math.h>

#define N_NODES 16384
#define CDIM    128
#define KTOP    20
#define RPB     32          // rows per block in the sim sweep (2 row tiles)

typedef __attribute__((ext_vector_type(16))) __bf16 v16bf;
typedef __attribute__((ext_vector_type(8)))  float  v8f;

__device__ __forceinline__ unsigned short f32_to_bf16(float f) {
  unsigned int u = __float_as_uint(f);
  u += 0x7fffu + ((u >> 16) & 1u);           // round-to-nearest-even
  return (unsigned short)(u >> 16);
}

// 16-bit A-matrix 16x32 fragment per CDNA5 ISA layout:
// lanes 0-15 : row = lane,    K = k0+{0..7} and k0+{16..23}
// lanes 16-31: row = lane-16, K = k0+{8..15} and k0+{24..31}
// B (32x16 column-major in memory == row-major of transposed array) mirrors it.
__device__ __forceinline__ v16bf load_frag(const unsigned short* base, int ld, int k0, int lane) {
  int half = (lane >> 4) & 1;
  int m = lane & 15;
  const unsigned short* p = base + (size_t)m * ld + k0 + half * 8;
  union { v16bf v; uint4 q[2]; } u;
  u.q[0] = *(const uint4*)(p);
  u.q[1] = *(const uint4*)(p + 16);
  return u.v;
}

__device__ __forceinline__ v8f wmma_bf16f32(v16bf a, v16bf b, v8f c) {
  return __builtin_amdgcn_wmma_f32_16x16x32_bf16(false, a, false, b, (short)0, c, false, false);
}

// ---------------------------------------------------------------------------
// Kernel 1: row L2-normalize x -> xn (bf16) and convert x -> xb (bf16).
// ---------------------------------------------------------------------------
__global__ __launch_bounds__(256) void normalize_kernel(const float* __restrict__ x,
                                                        unsigned short* __restrict__ xn,
                                                        unsigned short* __restrict__ xb) {
  int wave = threadIdx.x >> 5, lane = threadIdx.x & 31;
  int row = blockIdx.x * 8 + wave;
  const float4 v = ((const float4*)(x + (size_t)row * CDIM))[lane];
  float ss = v.x * v.x + v.y * v.y + v.z * v.z + v.w * v.w;
#pragma unroll
  for (int o = 16; o > 0; o >>= 1) ss += __shfl_xor(ss, o, 32);
  float inv = 1.0f / fmaxf(sqrtf(ss), 1e-8f);
  size_t o = (size_t)row * CDIM + lane * 4;
  xn[o + 0] = f32_to_bf16(v.x * inv);
  xn[o + 1] = f32_to_bf16(v.y * inv);
  xn[o + 2] = f32_to_bf16(v.z * inv);
  xn[o + 3] = f32_to_bf16(v.w * inv);
  xb[o + 0] = f32_to_bf16(v.x);
  xb[o + 1] = f32_to_bf16(v.y);
  xb[o + 2] = f32_to_bf16(v.z);
  xb[o + 3] = f32_to_bf16(v.w);
}

// ---------------------------------------------------------------------------
// Kernel 2: weight transpose slice -> bf16. dst[n][k] = src[k+off][n]
// ---------------------------------------------------------------------------
__global__ __launch_bounds__(256) void wprep_kernel(const float* __restrict__ src,
                                                    unsigned short* __restrict__ dst,
                                                    int k_off) {
  int t = blockIdx.x * 256 + threadIdx.x;   // 16384 = 128*128
  int n = t >> 7, k = t & 127;
  dst[n * CDIM + k] = f32_to_bf16(src[(size_t)(k + k_off) * CDIM + n]);
}

// ---------------------------------------------------------------------------
// Kernel 3: fused sim GEMM + streaming top-K + softmax.
// Block = 32 rows (2 A row-tiles in registers). 8 waves sweep 1024 column
// tiles with register double-buffered B fragments (8 WMMAs per B tile).
// Top-K lists live in LDS (not scratch!); the screening threshold is a VGPR.
// Lane <-> block-row mapping (row = lane) lets the final merge read the 8
// partial lists in place: row m's lists are at threads w*32+m.
// ---------------------------------------------------------------------------
__global__ __launch_bounds__(256) void simtopk_kernel(const unsigned short* __restrict__ xn,
                                                      int* __restrict__ idx,
                                                      float* __restrict__ score) {
  __shared__ __align__(16) float tileS[8][2][16][16];    // 16 KB
  __shared__ float ltv[256][KTOP + 1];                   // stride 21: no bank clash
  __shared__ int   lti[256][KTOP + 1];
  int tid = threadIdx.x;
  int wave = tid >> 5, lane = tid & 31;
  int half = lane >> 4, l16 = lane & 15;
  int r0 = blockIdx.x * RPB;
  int myrt = lane >> 4, myrow = lane & 15;               // scan row = lane

#pragma unroll
  for (int j = 0; j < KTOP; ++j) { ltv[tid][j] = -3.0e38f; lti[tid][j] = 0; }
  float thresh = -3.0e38f;                               // == ltv[tid][KTOP-1]

  // Two A row-tiles (32 rows x K=128) stay in registers for the whole sweep.
  v16bf afr[2][4];
#pragma unroll
  for (int rt = 0; rt < 2; ++rt)
#pragma unroll
    for (int ks = 0; ks < 4; ++ks)
      afr[rt][ks] = load_frag(xn + (size_t)(r0 + rt * 16) * CDIM, CDIM, ks * 32, lane);

  // rare path: ripple-insert into the LDS-resident sorted list
  auto insert = [&](float v, int c) {
    int p = KTOP - 1;
    while (p > 0 && ltv[tid][p - 1] < v) {
      ltv[tid][p] = ltv[tid][p - 1];
      lti[tid][p] = lti[tid][p - 1];
      --p;
    }
    ltv[tid][p] = v; lti[tid][p] = c;
    thresh = ltv[tid][KTOP - 1];
  };

  auto process = [&](int ct, const v16bf* b) {
    v8f acc0 = {0.f, 0.f, 0.f, 0.f, 0.f, 0.f, 0.f, 0.f};
    v8f acc1 = {0.f, 0.f, 0.f, 0.f, 0.f, 0.f, 0.f, 0.f};
#pragma unroll
    for (int ks = 0; ks < 4; ++ks) acc0 = wmma_bf16f32(afr[0][ks], b[ks], acc0);
#pragma unroll
    for (int ks = 0; ks < 4; ++ks) acc1 = wmma_bf16f32(afr[1][ks], b[ks], acc1);
#pragma unroll
    for (int g = 0; g < 8; ++g) tileS[wave][0][g + 8 * half][l16] = acc0[g];
#pragma unroll
    for (int g = 0; g < 8; ++g) tileS[wave][1][g + 8 * half][l16] = acc1[g];
    __builtin_amdgcn_wave_barrier();       // wave-private LDS tile; keep order
    int n0 = ct * 16;
    const float4* rowp = (const float4*)tileS[wave][myrt][myrow];
#pragma unroll
    for (int q = 0; q < 4; ++q) {
      float4 c = rowp[q];
      if (c.x > thresh) insert(c.x, n0 + q * 4 + 0);   // screen is all-register
      if (c.y > thresh) insert(c.y, n0 + q * 4 + 1);
      if (c.z > thresh) insert(c.z, n0 + q * 4 + 2);
      if (c.w > thresh) insert(c.w, n0 + q * 4 + 3);
    }
    __builtin_amdgcn_wave_barrier();
  };

  const int NTILES = N_NODES / 16;          // 1024 column tiles
  v16bf b0[4], b1[4];
#pragma unroll
  for (int ks = 0; ks < 4; ++ks)
    b0[ks] = load_frag(xn + (size_t)(wave * 16) * CDIM, CDIM, ks * 32, lane);

  for (int ct = wave; ct < NTILES; ct += 16) {
    int ctB = ct + 8, ctC = ct + 16;
    if (ctB < NTILES) {
      __builtin_prefetch(xn + (size_t)(ctC * 16) * CDIM, 0, 1);
#pragma unroll
      for (int ks = 0; ks < 4; ++ks)
        b1[ks] = load_frag(xn + (size_t)(ctB * 16) * CDIM, CDIM, ks * 32, lane);
    }
    process(ct, b0);                        // waits hidden behind b1 loads
    if (ctB < NTILES) {
      if (ctC < NTILES) {
#pragma unroll
        for (int ks = 0; ks < 4; ++ks)
          b0[ks] = load_frag(xn + (size_t)(ctC * 16) * CDIM, CDIM, ks * 32, lane);
      }
      process(ctB, b1);
    }
  }

  __syncthreads();
  // merge: 8 sorted partial lists per row -> final top-20 -> softmax
  if (tid < RPB) {
    int m = tid;
    float fv[KTOP]; int fi[KTOP];
#pragma unroll
    for (int j = 0; j < KTOP; ++j) { fv[j] = -3.0e38f; fi[j] = 0; }
    for (int w = 0; w < 8; ++w) {
      const float* pv = ltv[w * 32 + m];
      const int*   pi = lti[w * 32 + m];
      for (int j = 0; j < KTOP; ++j) {
        float v = pv[j];
        if (v <= fv[KTOP - 1]) break;       // lists sorted descending
        int p = KTOP - 1;
        while (p > 0 && fv[p - 1] < v) { fv[p] = fv[p - 1]; fi[p] = fi[p - 1]; --p; }
        fv[p] = v; fi[p] = pi[j];
      }
    }
    float mx = fv[0], s = 0.f, e[KTOP];
    for (int j = 0; j < KTOP; ++j) { e[j] = __expf(fv[j] - mx); s += e[j]; }
    float invs = 1.0f / s;
    size_t o = (size_t)(r0 + m) * KTOP;
    for (int j = 0; j < KTOP; ++j) { score[o + j] = e[j] * invs; idx[o + j] = fi[j]; }
  }
}

// ---------------------------------------------------------------------------
// Kernel 4: generic [N,128]@[128,128] bf16 WMMA GEMM with fused epilogue:
// out = maybe_relu(A@W + bias + addend) + residual -> f32 and/or bf16.
// ---------------------------------------------------------------------------
__global__ __launch_bounds__(256) void gemm128_kernel(
    const unsigned short* __restrict__ A,
    const unsigned short* __restrict__ WT,
    const float* __restrict__ bias,
    const float* __restrict__ addend,
    const float* __restrict__ residual,
    int do_relu,
    float* __restrict__ outF,
    unsigned short* __restrict__ outBF) {
  int wave = threadIdx.x >> 5, lane = threadIdx.x & 31;
  int half = lane >> 4, l16 = lane & 15;
  int r0 = blockIdx.x * 16, n0 = wave * 16;
  v16bf af[4], bf[4];
#pragma unroll
  for (int ks = 0; ks < 4; ++ks) af[ks] = load_frag(A + (size_t)r0 * CDIM, CDIM, ks * 32, lane);
#pragma unroll
  for (int ks = 0; ks < 4; ++ks) bf[ks] = load_frag(WT + (size_t)n0 * CDIM, CDIM, ks * 32, lane);
  v8f acc = {0.f, 0.f, 0.f, 0.f, 0.f, 0.f, 0.f, 0.f};
#pragma unroll
  for (int ks = 0; ks < 4; ++ks) acc = wmma_bf16f32(af[ks], bf[ks], acc);
  int n = n0 + l16;
  float bv = bias ? bias[n] : 0.0f;
#pragma unroll
  for (int g = 0; g < 8; ++g) {
    int row = r0 + g + 8 * half;
    size_t o = (size_t)row * CDIM + n;
    float v = acc[g] + bv;
    if (addend)   v += addend[o];
    if (do_relu)  v = fmaxf(v, 0.0f);
    if (residual) v += residual[o];
    if (outF)  outF[o] = v;
    if (outBF) outBF[o] = f32_to_bf16(v);
  }
}

// ---------------------------------------------------------------------------
// Kernel 5: per-node edge message MLP + weighted aggregation.
// H1 = relu(T_i + Xj@W1_bot)  [32,128] bf16 in LDS
// h_agg[i] = sum_m score[i][m] * (H1@W2)[m] + msg_b2  (softmax sums to 1)
// ---------------------------------------------------------------------------
__global__ __launch_bounds__(256) void message_kernel(
    const unsigned short* __restrict__ xb,
    const int* __restrict__ idx,
    const float* __restrict__ score,
    const float* __restrict__ Tmat,
    const unsigned short* __restrict__ W1bT,
    const unsigned short* __restrict__ W2T,
    const float* __restrict__ b2,
    float* __restrict__ hagg,
    unsigned short* __restrict__ haggBF) {
  __shared__ __align__(16) unsigned short Xj[32 * CDIM];
  __shared__ __align__(16) unsigned short H1[32 * CDIM];
  __shared__ float Ti[CDIM];
  int i = blockIdx.x, tid = threadIdx.x;
  if (tid < CDIM) Ti[tid] = Tmat[(size_t)i * CDIM + tid];
  {
    int row = tid >> 3, c0 = (tid & 7) * 16;       // 256 thr x 16 elems = 32x128
    if (row < KTOP) {
      const unsigned short* src = xb + (size_t)idx[i * KTOP + row] * CDIM + c0;
      *(uint4*)&Xj[row * CDIM + c0]     = *(const uint4*)(src);
      *(uint4*)&Xj[row * CDIM + c0 + 8] = *(const uint4*)(src + 8);
    } else {
      uint4 z = make_uint4(0, 0, 0, 0);
      *(uint4*)&Xj[row * CDIM + c0]     = z;
      *(uint4*)&Xj[row * CDIM + c0 + 8] = z;
    }
  }
  __syncthreads();
  int wave = tid >> 5, lane = tid & 31, half = lane >> 4, l16 = lane & 15;
  int n0 = wave * 16, n = n0 + l16;

  v16bf wf[4];
#pragma unroll
  for (int ks = 0; ks < 4; ++ks) wf[ks] = load_frag(W1bT + (size_t)n0 * CDIM, CDIM, ks * 32, lane);
#pragma unroll
  for (int rt = 0; rt < 2; ++rt) {
    v16bf af[4];
#pragma unroll
    for (int ks = 0; ks < 4; ++ks) af[ks] = load_frag(&Xj[rt * 16 * CDIM], CDIM, ks * 32, lane);
    v8f acc = {0.f, 0.f, 0.f, 0.f, 0.f, 0.f, 0.f, 0.f};
#pragma unroll
    for (int ks = 0; ks < 4; ++ks) acc = wmma_bf16f32(af[ks], wf[ks], acc);
    float tval = Ti[n];
#pragma unroll
    for (int g = 0; g < 8; ++g) {
      int m = rt * 16 + g + 8 * half;
      H1[m * CDIM + n] = f32_to_bf16(fmaxf(acc[g] + tval, 0.0f));
    }
  }
  __syncthreads();

  float colsum = 0.0f;
#pragma unroll
  for (int ks = 0; ks < 4; ++ks) wf[ks] = load_frag(W2T + (size_t)n0 * CDIM, CDIM, ks * 32, lane);
#pragma unroll
  for (int rt = 0; rt < 2; ++rt) {
    v16bf af[4];
#pragma unroll
    for (int ks = 0; ks < 4; ++ks) af[ks] = load_frag(&H1[rt * 16 * CDIM], CDIM, ks * 32, lane);
    v8f acc = {0.f, 0.f, 0.f, 0.f, 0.f, 0.f, 0.f, 0.f};
#pragma unroll
    for (int ks = 0; ks < 4; ++ks) acc = wmma_bf16f32(af[ks], wf[ks], acc);
#pragma unroll
    for (int g = 0; g < 8; ++g) {
      int m = rt * 16 + g + 8 * half;
      float s = (m < KTOP) ? score[i * KTOP + m] : 0.0f;   // pad rows weight 0
      colsum += s * acc[g];
    }
  }
  colsum += __shfl_xor(colsum, 16, 32);   // join the two row-halves per column
  if (lane < 16) {
    float v = colsum + b2[n];
    size_t o = (size_t)i * CDIM + n;
    hagg[o]   = v;
    haggBF[o] = f32_to_bf16(v);
  }
}

// ---------------------------------------------------------------------------
extern "C" void kernel_launch(void* const* d_in, const int* in_sizes, int n_in,
                              void* d_out, int out_size, void* d_ws, size_t ws_size,
                              hipStream_t stream) {
  (void)in_sizes; (void)n_in; (void)out_size; (void)ws_size;
  const float* x      = (const float*)d_in[0];
  const float* msg_w1 = (const float*)d_in[1];
  const float* msg_b1 = (const float*)d_in[2];
  const float* msg_w2 = (const float*)d_in[3];
  const float* msg_b2 = (const float*)d_in[4];
  const float* upd_w1 = (const float*)d_in[5];
  const float* upd_b1 = (const float*)d_in[6];
  const float* upd_w2 = (const float*)d_in[7];
  const float* upd_b2 = (const float*)d_in[8];

  const size_t NC = (size_t)N_NODES * CDIM;
  char* ws = (char*)d_ws;
  size_t off = 0;
  auto take = [&](size_t bytes) -> void* {
    void* p = (void*)(ws + off);
    off += (bytes + 255) & ~(size_t)255;
    return p;
  };
  unsigned short* xn_bf   = (unsigned short*)take(NC * 2);
  unsigned short* xb_bf   = (unsigned short*)take(NC * 2);
  unsigned short* w_m1t   = (unsigned short*)take(CDIM * CDIM * 2);
  unsigned short* w_m1b   = (unsigned short*)take(CDIM * CDIM * 2);
  unsigned short* w_m2    = (unsigned short*)take(CDIM * CDIM * 2);
  unsigned short* w_u1t   = (unsigned short*)take(CDIM * CDIM * 2);
  unsigned short* w_u1b   = (unsigned short*)take(CDIM * CDIM * 2);
  unsigned short* w_u2    = (unsigned short*)take(CDIM * CDIM * 2);
  float*          Tmat    = (float*)take(NC * 4);
  int*            idxb    = (int*)take((size_t)N_NODES * KTOP * 4);
  float*          scoreb  = (float*)take((size_t)N_NODES * KTOP * 4);
  float*          hagg    = (float*)take(NC * 4);
  unsigned short* hagg_bf = (unsigned short*)take(NC * 2);
  float*          Umat    = (float*)take(NC * 4);
  unsigned short* h2_bf   = (unsigned short*)take(NC * 2);

  normalize_kernel<<<N_NODES / 8, 256, 0, stream>>>(x, xn_bf, xb_bf);
  wprep_kernel<<<64, 256, 0, stream>>>(msg_w1, w_m1t, 0);
  wprep_kernel<<<64, 256, 0, stream>>>(msg_w1, w_m1b, CDIM);
  wprep_kernel<<<64, 256, 0, stream>>>(msg_w2, w_m2, 0);
  wprep_kernel<<<64, 256, 0, stream>>>(upd_w1, w_u1t, 0);
  wprep_kernel<<<64, 256, 0, stream>>>(upd_w1, w_u1b, CDIM);
  wprep_kernel<<<64, 256, 0, stream>>>(upd_w2, w_u2, 0);

  // Fused cosine-sim GEMM + top-K + softmax (the 68.7 GFLOP hot loop).
  simtopk_kernel<<<N_NODES / RPB, 256, 0, stream>>>(xn_bf, idxb, scoreb);

  // T = x @ msg_W1_top + msg_b1 (shared across all K edges of each node).
  gemm128_kernel<<<N_NODES / 16, 256, 0, stream>>>(xb_bf, w_m1t, msg_b1,
                                                   nullptr, nullptr, 0, Tmat, nullptr);
  // Per-edge message MLP + weighted scatter-sum.
  message_kernel<<<N_NODES, 256, 0, stream>>>(xb_bf, idxb, scoreb, Tmat,
                                              w_m1b, w_m2, msg_b2, hagg, hagg_bf);
  // Update MLP: U = x@U1top + b1 ; H2 = relu(U + hagg@U1bot) ; out = H2@U2 + b2 + x
  gemm128_kernel<<<N_NODES / 16, 256, 0, stream>>>(xb_bf, w_u1t, upd_b1,
                                                   nullptr, nullptr, 0, Umat, nullptr);
  gemm128_kernel<<<N_NODES / 16, 256, 0, stream>>>(hagg_bf, w_u1b, nullptr,
                                                   Umat, nullptr, 1, nullptr, h2_bf);
  gemm128_kernel<<<N_NODES / 16, 256, 0, stream>>>(h2_bf, w_u2, upd_b2,
                                                   nullptr, x, 0, (float*)d_out, nullptr);
}
